// BasisCustBiLSTM_23888608100430
// MI455X (gfx1250) — matrix operand order repaired and checked by reading, not compiled, not run on hardware
//
#include <hip/hip_runtime.h>
#include <hip/hip_bf16.h>
#include <math.h>

// ---------------------------------------------------------------------------
// BasisCustBiLSTM for MI455X (gfx1250, wave32, WMMA).
// B=32, T=256, I=CELL=512, NB=8, EMB=64, KQ=64, gates 4C=2048.
//
//  * Per-sample mixed weights materialized once in f16 (Whh fwd+rev = 134MB
//    -> resident in MI455X 192MB L2; scan streams them from L2 every step).
//  * x-dependent gates hoisted into one WMMA f16 GEMM (time-parallel).
//  * 256-step scan: 64 blocks (sample x dir). Per step:
//      - h B-fragments hoisted to registers once (kills redundant DS traffic)
//      - gates_x[s+1] staged into LDS with global_load_async_to_lds_b128,
//        overlapped with the WMMA phase, ordered via s_wait_asynccnt.
// NOTE: mask (input 1) assumed 1-byte bool (jnp.bool_).
// ---------------------------------------------------------------------------

typedef __attribute__((ext_vector_type(16))) _Float16 v16h;
typedef __attribute__((ext_vector_type(8)))  _Float16 h8;
typedef __attribute__((ext_vector_type(4)))  _Float16 h4;
typedef __attribute__((ext_vector_type(8)))  float    v8f;
typedef __attribute__((ext_vector_type(4)))  float    f4;

#define BB   32
#define TT   256
#define II   512
#define CC   512
#define GG   2048      // 4*CELL gate rows
#define NBAS 8
#define EMBD 64
#define KQD  64

__device__ __forceinline__ float fast_tanh(float x) {
#if defined(__has_builtin)
#if __has_builtin(__builtin_amdgcn_tanhf)
  return __builtin_amdgcn_tanhf(x);
#else
  return tanhf(x);
#endif
#else
  return tanhf(x);
#endif
}

__device__ __forceinline__ float fast_sigmoid(float x) {
  return 1.f / (1.f + __expf(-x));
}

// ---------------------------------------------------------------------------
// K1: mixture coefficients c_batch[b,8] + lengths[b]. One block per sample.
// ---------------------------------------------------------------------------
__global__ void __launch_bounds__(128)
k_cbatch(const unsigned char* __restrict__ mask,
         const int* __restrict__ ma, const int* __restrict__ mc,
         const float* __restrict__ ea, const float* __restrict__ ec,
         const float* __restrict__ W1, const float* __restrict__ b1,
         const float* __restrict__ W2,
         float* __restrict__ cb, int* __restrict__ len) {
  const int b = blockIdx.x, tid = threadIdx.x;
  __shared__ float q[2 * EMBD];
  __shared__ float hid[KQD];
  __shared__ float red[128];
  __shared__ float lg[NBAS];

  if (tid < EMBD) q[tid] = ea[ma[b] * EMBD + tid];
  else            q[tid] = ec[mc[b] * EMBD + (tid - EMBD)];

  int s = (int)(mask[b * TT + tid] != 0) + (int)(mask[b * TT + 128 + tid] != 0);
  red[tid] = (float)s;
  __syncthreads();
  for (int st = 64; st > 0; st >>= 1) {
    if (tid < st) red[tid] += red[tid + st];
    __syncthreads();
  }
  if (tid == 0) len[b] = (int)red[0];

  if (tid < KQD) {
    float acc = b1[tid];
    #pragma unroll 8
    for (int k = 0; k < 2 * EMBD; ++k) acc += q[k] * W1[k * KQD + tid];
    hid[tid] = fast_tanh(acc);
  }
  __syncthreads();
  if (tid < NBAS) {
    float acc = 0.f;
    #pragma unroll 8
    for (int k = 0; k < KQD; ++k) acc += hid[k] * W2[k * NBAS + tid];
    lg[tid] = acc;
  }
  __syncthreads();
  if (tid == 0) {
    float m = lg[0];
    for (int n = 1; n < NBAS; ++n) m = fmaxf(m, lg[n]);
    float e[NBAS], ssum = 0.f;
    for (int n = 0; n < NBAS; ++n) { e[n] = __expf(lg[n] - m); ssum += e[n]; }
    for (int n = 0; n < NBAS; ++n) cb[b * NBAS + n] = e[n] / ssum;
  }
}

// ---------------------------------------------------------------------------
// K2a: x f32 -> f16 (WMMA A operand source). 4 elems/thread.
// ---------------------------------------------------------------------------
__global__ void __launch_bounds__(256)
k_x16(const float* __restrict__ x, _Float16* __restrict__ o) {
  const long i = (long)(blockIdx.x * blockDim.x + threadIdx.x) * 4;
  const f4 v = *(const f4*)(x + i);
  h4 h;
  h.x = (_Float16)v.x; h.y = (_Float16)v.y;
  h.z = (_Float16)v.z; h.w = (_Float16)v.w;
  *(h4*)(o + i) = h;
}

// ---------------------------------------------------------------------------
// K2b: mix one weight tensor: out16[b,g,k] = sum_n c[b,n]*W[n,g,k].
// ---------------------------------------------------------------------------
__global__ void __launch_bounds__(256)
k_mixw(const float* __restrict__ W, const float* __restrict__ cb,
       _Float16* __restrict__ out) {
  const int  b    = blockIdx.y;
  const long base = (long)(blockIdx.x * blockDim.x + threadIdx.x) * 8;
  float c[NBAS];
  #pragma unroll
  for (int n = 0; n < NBAS; ++n) c[n] = cb[b * NBAS + n];
  float acc[8] = {0.f, 0.f, 0.f, 0.f, 0.f, 0.f, 0.f, 0.f};
  #pragma unroll
  for (int n = 0; n < NBAS; ++n) {
    const f4 w0 = *(const f4*)(W + (long)n * GG * II + base);
    const f4 w1 = *(const f4*)(W + (long)n * GG * II + base + 4);
    acc[0] += c[n] * w0.x; acc[1] += c[n] * w0.y;
    acc[2] += c[n] * w0.z; acc[3] += c[n] * w0.w;
    acc[4] += c[n] * w1.x; acc[5] += c[n] * w1.y;
    acc[6] += c[n] * w1.z; acc[7] += c[n] * w1.w;
  }
  h8 hv;
  #pragma unroll
  for (int e = 0; e < 8; ++e) hv[e] = (_Float16)acc[e];
  *(h8*)(out + (long)b * GG * II + base) = hv;
}

// ---------------------------------------------------------------------------
// K2c: mix biases: bmix[b,g] = sum_n c[b,n]*bias[n,g].
// ---------------------------------------------------------------------------
__global__ void __launch_bounds__(256)
k_mixbias(const float* __restrict__ bias, const float* __restrict__ cb,
          float* __restrict__ bmix) {
  const int i = blockIdx.x * blockDim.x + threadIdx.x; // 0..B*GG-1
  const int b = i >> 11, g = i & (GG - 1);
  float acc = 0.f;
  #pragma unroll
  for (int n = 0; n < NBAS; ++n) acc += cb[b * NBAS + n] * bias[n * GG + g];
  bmix[i] = acc;
}

// ---------------------------------------------------------------------------
// K3: gates_x[b,t,g] = x[b,t(or T-1-t),:] @ Wih16[b]^T + bmix[b]  (WMMA GEMM)
// One 16x16 output tile per wave; K = 512 in 32-chunks.
// ---------------------------------------------------------------------------
__global__ void __launch_bounds__(256)
k_gatesx(const _Float16* __restrict__ x16,
         const _Float16* __restrict__ WihF, const _Float16* __restrict__ WihR,
         const float* __restrict__ bmixF,  const float* __restrict__ bmixR,
         float* __restrict__ gxF,          float* __restrict__ gxR) {
  const int lane = threadIdx.x & 31;
  const int wave = threadIdx.x >> 5;
  int idx = blockIdx.x * 8 + wave;
  const int gT = idx & 127; idx >>= 7;
  const int tT = idx & 15;  idx >>= 4;
  const int b  = idx & 31;  idx >>= 5;
  const int d  = idx;

  const int lo = (lane < 16);
  const int m  = lane & 15;
  const int trowBase = tT * 16 + m;
  const int trow = d ? (TT - 1 - trowBase) : trowBase;

  const _Float16* __restrict__ Xrow = x16 + ((long)b * TT + trow) * II;
  const _Float16* __restrict__ W16  =
      (d ? WihR : WihF) + (long)b * GG * II + (long)(gT * 16 + m) * II;

  v8f acc = {};
  #pragma unroll 4
  for (int k = 0; k < II; k += 32) {
    union { v16h v; h8 h[2]; } ua;   // A: 16x32 f16 tile of x (ISA layout)
    ua.h[0] = *(const h8*)(Xrow + k + (lo ? 0 : 8));
    ua.h[1] = *(const h8*)(Xrow + k + (lo ? 16 : 24));
    union { v16h v; h8 h[2]; } ubm;  // B: col = W row g
    ubm.h[0] = *(const h8*)(W16 + k + (lo ? 0 : 8));
    ubm.h[1] = *(const h8*)(W16 + k + (lo ? 16 : 24));
    const v16h bm = ubm.v;
    acc = __builtin_amdgcn_wmma_f32_16x16x32_f16(false, ua.v, false, bm,
                                                 (short)0, acc, false, false);
  }

  const float* __restrict__ bmix = d ? bmixR : bmixF;
  const int n = lane & 15;
  const float bias = bmix[b * GG + gT * 16 + n];
  float* __restrict__ gx =
      (d ? gxR : gxF) + ((long)b * TT + tT * 16) * GG + gT * 16 + n;
  const int mhi = lo ? 0 : 8;
  #pragma unroll
  for (int j = 0; j < 8; ++j)
    gx[(long)(j + mhi) * GG] = acc[j] + bias;
}

// ---------------------------------------------------------------------------
// K4: sequential scan. One block per (sample, dir). Per step:
//   - async-stage gates_x[s+1] into LDS (overlapped with WMMA phase)
//   - gacc = Whh16[b] @ h via WMMA (h fragments hoisted to registers)
//   - LSTM cell update, validity masking, write h to out.
// ---------------------------------------------------------------------------
__global__ void __launch_bounds__(256)
k_scan(const _Float16* __restrict__ WhhF, const _Float16* __restrict__ WhhR,
       const float* __restrict__ gxF,     const float* __restrict__ gxR,
       const int* __restrict__ len,       float* __restrict__ out) {
  const int b    = blockIdx.x & 31;
  const int d    = blockIdx.x >> 5;
  const int tid  = threadIdx.x;
  const int lane = tid & 31;
  const int wave = tid >> 5;
  const int lo   = (lane < 16);

  __shared__ _Float16 h16[CC];
  __shared__ float    gacc[GG];
  __shared__ float    shGX[2][GG];   // double-buffered gates_x staging

  const _Float16* __restrict__ Whh = (d ? WhhR : WhhF) + (long)b * GG * CC;
  const float*    __restrict__ gx  = (d ? gxR  : gxF ) + (long)b * TT * GG;
  const int L = len[b];

  float c0 = 0.f, c1 = 0.f;
  h16[tid]       = (_Float16)0.f;
  h16[tid + 256] = (_Float16)0.f;

  // Prologue: async-stage gates_x[0] into buffer 0 (2 x b128 per thread).
  {
    const unsigned dst =
        (unsigned)(uintptr_t)(&shGX[0][tid * 8]);
    const unsigned long long src =
        (unsigned long long)(uintptr_t)(gx + tid * 8);
    asm volatile(
        "global_load_async_to_lds_b128 %0, %1, off\n\t"
        "global_load_async_to_lds_b128 %0, %1, off offset:16"
        :: "v"(dst), "v"(src) : "memory");
  }
  __syncthreads();

  for (int s = 0; s < TT; ++s) {
    // ---- issue async stage of gates_x[s+1] (wraps at the end so the
    //      per-wave ASYNCcnt discipline stays constant) ----
    {
      const int sn = (s + 1 < TT) ? (s + 1) : 0;
      const unsigned dst =
          (unsigned)(uintptr_t)(&shGX[sn & 1][tid * 8]);
      const unsigned long long src =
          (unsigned long long)(uintptr_t)(gx + (long)sn * GG + tid * 8);
      asm volatile(
          "global_load_async_to_lds_b128 %0, %1, off\n\t"
          "global_load_async_to_lds_b128 %0, %1, off offset:16"
          :: "v"(dst), "v"(src) : "memory");
    }

    // ---- hoist h B-fragments (replicated across N cols) to registers ----
    v16h bfrag[16];
    #pragma unroll
    for (int kk = 0; kk < 16; ++kk) {
      union { v16h v; h8 h[2]; } ub;
      const int kb = kk * 32 + (lo ? 0 : 8);
      ub.h[0] = *(const h8*)(&h16[kb]);
      ub.h[1] = *(const h8*)(&h16[kb + 16]);
      bfrag[kk] = ub.v;
    }

    // ---- matrix phase: gacc = Whh @ h (128 row-tiles over 8 waves) ----
    #pragma unroll 1
    for (int i = 0; i < 16; ++i) {
      const int tile = i * 8 + wave;
      const _Float16* __restrict__ Wrow =
          Whh + (long)(tile * 16 + (lane & 15)) * CC;
      v8f acc = {};
      #pragma unroll 4
      for (int kk = 0; kk < 16; ++kk) {
        union { v16h v; h8 h[2]; } ua;          // A: Whh 16x32 tile
        ua.h[0] = *(const h8*)(Wrow + kk * 32 + (lo ? 0 : 8));
        ua.h[1] = *(const h8*)(Wrow + kk * 32 + (lo ? 16 : 24));
        acc = __builtin_amdgcn_wmma_f32_16x16x32_f16(
            false, ua.v, false, bfrag[kk], (short)0, acc, false, false);
      }
      if ((lane & 15) == 0) {                    // all N cols equal; one writer
        const int mbase = tile * 16 + (lo ? 0 : 8);
        #pragma unroll
        for (int j = 0; j < 8; ++j) gacc[mbase + j] = acc[j];
      }
    }

    // The 2 async instrs for step s (issued last iteration / prologue) have
    // completed once ASYNCcnt <= 2 (in-order completion; the 2 just-issued
    // for s+1 may still be in flight).
    asm volatile("s_wait_asynccnt 0x2" ::: "memory");
    __syncthreads();

    // ---- elementwise cell update: 2 cells per thread ----
    const float* __restrict__ g = shGX[s & 1];
    const bool valid = d ? ((TT - s) <= L) : (s < L);
    float hv0, hv1;
    {
      const int cell = tid;
      const float ig = g[cell]        + gacc[cell];
      const float fg = g[cell + 512]  + gacc[cell + 512];
      const float gg = g[cell + 1024] + gacc[cell + 1024];
      const float og = g[cell + 1536] + gacc[cell + 1536];
      float cn = fast_sigmoid(fg) * c0 + fast_sigmoid(ig) * fast_tanh(gg);
      float hn = fast_sigmoid(og) * fast_tanh(cn);
      c0  = valid ? cn : 0.f;
      hv0 = valid ? hn : 0.f;
    }
    {
      const int cell = tid + 256;
      const float ig = g[cell]        + gacc[cell];
      const float fg = g[cell + 512]  + gacc[cell + 512];
      const float gg = g[cell + 1024] + gacc[cell + 1024];
      const float og = g[cell + 1536] + gacc[cell + 1536];
      float cn = fast_sigmoid(fg) * c1 + fast_sigmoid(ig) * fast_tanh(gg);
      float hn = fast_sigmoid(og) * fast_tanh(cn);
      c1  = valid ? cn : 0.f;
      hv1 = valid ? hn : 0.f;
    }
    h16[tid]       = (_Float16)hv0;
    h16[tid + 256] = (_Float16)hv1;

    const int t = d ? (TT - 1 - s) : s;
    float* __restrict__ op = out + ((long)b * TT + t) * (2 * CC) + (d ? CC : 0);
    op[tid]       = hv0;
    op[tid + 256] = hv1;
    __syncthreads();
  }
}

// ---------------------------------------------------------------------------
// Host-side launch
// ---------------------------------------------------------------------------
extern "C" void kernel_launch(void* const* d_in, const int* in_sizes, int n_in,
                              void* d_out, int out_size, void* d_ws, size_t ws_size,
                              hipStream_t stream) {
  (void)in_sizes; (void)n_in; (void)out_size; (void)ws_size;

  const float*         x    = (const float*)d_in[0];
  const unsigned char* mask = (const unsigned char*)d_in[1]; // jnp bool_
  const int*   ma   = (const int*)d_in[2];
  const int*   mc   = (const int*)d_in[3];
  const float* ea   = (const float*)d_in[4];
  const float* ec   = (const float*)d_in[5];
  const float* W1   = (const float*)d_in[6];
  const float* b1   = (const float*)d_in[7];
  const float* W2   = (const float*)d_in[8];
  const float* Wih  = (const float*)d_in[9];
  const float* Whh  = (const float*)d_in[10];
  const float* bi   = (const float*)d_in[11];
  const float* WihRv= (const float*)d_in[12];
  const float* WhhRv= (const float*)d_in[13];
  const float* biRv = (const float*)d_in[14];
  float* out = (float*)d_out;
  char*  ws  = (char*)d_ws;

  constexpr size_t SZ_C    = (size_t)BB * NBAS * 4;
  constexpr size_t SZ_LEN  = (size_t)BB * 4;
  constexpr size_t SZ_BMIX = (size_t)BB * GG * 4;
  constexpr size_t SZ_X16  = (size_t)BB * TT * II * 2;
  constexpr size_t SZ_W16  = (size_t)BB * GG * II * 2;
  constexpr size_t SZ_GX   = (size_t)BB * TT * GG * 4;

  size_t off = 0;
  auto take = [&](size_t sz) { size_t o = off; off += (sz + 255) & ~(size_t)255; return o; };
  const size_t oC    = take(SZ_C);
  const size_t oLEN  = take(SZ_LEN);
  const size_t oBMF  = take(SZ_BMIX);
  const size_t oBMR  = take(SZ_BMIX);
  const size_t oX16  = take(SZ_X16);
  const size_t oWIF  = take(SZ_W16);
  const size_t oWIR  = take(SZ_W16);
  const size_t oWHF  = take(SZ_W16);
  const size_t oWHR  = take(SZ_W16);
  const size_t oGXF  = take(SZ_GX);
  const size_t oGXR  = take(SZ_GX);

  float*    cb    = (float*)(ws + oC);
  int*      len   = (int*)  (ws + oLEN);
  float*    bmixF = (float*)(ws + oBMF);
  float*    bmixR = (float*)(ws + oBMR);
  _Float16* x16   = (_Float16*)(ws + oX16);
  _Float16* WihF16= (_Float16*)(ws + oWIF);
  _Float16* WihR16= (_Float16*)(ws + oWIR);
  _Float16* WhhF16= (_Float16*)(ws + oWHF);
  _Float16* WhhR16= (_Float16*)(ws + oWHR);
  float*    gxF   = (float*)(ws + oGXF);
  float*    gxR   = (float*)(ws + oGXR);

  k_cbatch<<<BB, 128, 0, stream>>>(mask, ma, mc, ea, ec, W1, b1, W2, cb, len);
  k_x16<<<(BB * TT * II) / (256 * 4), 256, 0, stream>>>(x, x16);

  const dim3 gmw((GG * II) / (256 * 8), BB);
  k_mixw<<<gmw, 256, 0, stream>>>(Wih,   cb, WihF16);
  k_mixw<<<gmw, 256, 0, stream>>>(Whh,   cb, WhhF16);
  k_mixw<<<gmw, 256, 0, stream>>>(WihRv, cb, WihR16);
  k_mixw<<<gmw, 256, 0, stream>>>(WhhRv, cb, WhhR16);
  k_mixbias<<<(BB * GG) / 256, 256, 0, stream>>>(bi,   cb, bmixF);
  k_mixbias<<<(BB * GG) / 256, 256, 0, stream>>>(biRv, cb, bmixR);

  k_gatesx<<<(2 * BB * (TT / 16) * (GG / 16)) / 8, 256, 0, stream>>>(
      x16, WihF16, WihR16, bmixF, bmixR, gxF, gxR);

  k_scan<<<2 * BB, 256, 0, stream>>>(WhhF16, WhhR16, gxF, gxR, len, out);
}
